// MessagePassing_59339268162203
// MI455X (gfx1250) — matrix operand order repaired and checked by reading, not compiled, not run on hardware
//
#include <hip/hip_runtime.h>
#include <stdint.h>

// ---------------------------------------------------------------------------
// GNN message-passing block for MI455X (gfx1250).
// Dense 128x128 GEMMs via v_wmma_f32_16x16x32_bf16 (f32 accumulate),
// sparse segment-sum via per-edge wave + inline-asm global_atomic_add_f32.
// ---------------------------------------------------------------------------

constexpr int DIM   = 128;   // feature dim
constexpr int TROWS = 64;    // row tile per block
constexpr int LDAP  = 136;   // LDS row stride (bf16 elems), avoids bank conflicts

typedef __attribute__((ext_vector_type(16))) __bf16   v16bf;
typedef __attribute__((ext_vector_type(8)))  float    v8f;
typedef __attribute__((ext_vector_type(4)))  unsigned u32x4;
typedef __attribute__((ext_vector_type(2)))  unsigned u32x2;
typedef __attribute__((ext_vector_type(4)))  float    f32x4;

union Frag {
  v16bf v;
  u32x4 q[2];
};

__device__ __forceinline__ unsigned short f2bf(float f) {
  union { float f; uint32_t u; } c; c.f = f;
  uint32_t u = c.u;
  return (unsigned short)((u + 0x7FFFu + ((u >> 16) & 1u)) >> 16);  // RNE
}

__device__ __forceinline__ u32x2 pack4bf(f32x4 v) {
  u32x2 r;
  r.x = (unsigned)f2bf(v.x) | ((unsigned)f2bf(v.y) << 16);
  r.y = (unsigned)f2bf(v.z) | ((unsigned)f2bf(v.w) << 16);
  return r;
}

__device__ __forceinline__ float sigmoidf_(float v) {
  return 1.0f / (1.0f + __expf(-v));
}

// Hardware f32 atomic add (no return): single VMEM op, STOREcnt-tracked;
// drained by the implicit wait-idle at s_endpgm.
__device__ __forceinline__ void gfadd(float* p, float v) {
  asm volatile("global_atomic_add_f32 %0, %1, off"
               :
               : "v"((unsigned long long)(uintptr_t)p), "v"(v)
               : "memory");
}

// Stage a 64x128 f32 tile from global into LDS as bf16 (row stride LDAP).
// float4 loads -> packed bf16 ds_store_b64.
__device__ __forceinline__ void stage_tile(const float* __restrict__ src,
                                           unsigned short* lds, int row0, int n) {
  for (int idx = threadIdx.x; idx < TROWS * (DIM / 4); idx += 128) {
    const int r  = idx >> 5;           // DIM/4 == 32 chunks per row
    const int c4 = (idx & 31) * 4;
    int gr = row0 + r; if (gr >= n) gr = n - 1;   // clamp (stores are guarded)
    const f32x4 v = *(const f32x4*)(src + (size_t)gr * DIM + c4);
    *(u32x2*)(lds + r * LDAP + c4) = pack4bf(v);
  }
}

// One wave computes a 16x128 strip of  tile(64x128) @ W(128x128).
// lds: bf16 tile, row-major, row stride LDAP.
// wt : bf16 transposed weights, wt[n*DIM + k] = W[k][n] (column of W contiguous).
// acc: 8 f32 16x16 accumulators (caller init / chained across calls).
__device__ __forceinline__ void gemm16x128(const unsigned short* lds, int row16,
                                           const unsigned short* __restrict__ wt,
                                           v8f acc[8]) {
  const int lane = threadIdx.x & 31;
  const int mrow = row16 + (lane & 15);
  const int akb  = (lane < 16) ? 0 : 8;   // A: K sub-offset per half-wave
  const int bkb  = (lane < 16) ? 0 : 8;   // B: K sub-offset per half-wave
  const int ncol = lane & 15;
#pragma unroll
  for (int kt = 0; kt < 4; ++kt) {        // K = 128 in steps of 32
    Frag a;
    // A 16x32 bf16 layout: elems 0..7 -> K = kt*32+akb+j ; elems 8..15 -> +16
    const u32x4* ap = (const u32x4*)(lds + mrow * LDAP + kt * 32 + akb);
    a.q[0] = ap[0];
    a.q[1] = ap[2];  // +32 bytes = +16 elems
#pragma unroll
    for (int nt = 0; nt < 8; ++nt) {      // N = 128 in steps of 16
      Frag b;
      // B 32x16 layout: lane holds column n, elems 0..7 -> K = kt*32+bkb+j ; elems 8..15 -> +16
      const u32x4* bp = (const u32x4*)(wt + (nt * 16 + ncol) * DIM + kt * 32 + bkb);
      b.q[0] = bp[0];
      b.q[1] = bp[2];
      acc[nt] = __builtin_amdgcn_wmma_f32_16x16x32_bf16(
          false, a.v, false, b.v, (short)0, acc[nt], false, false);
    }
  }
}

// ---------------- weight conversion: f32 [k][n] -> bf16 transposed [n][k] ----
struct WPtrs { const float* p[10]; };

__global__ void wcvt_kernel(WPtrs w, unsigned short* __restrict__ dst) {
  const int mat = blockIdx.y;
  const float* __restrict__ s = w.p[mat];
  unsigned short* __restrict__ d = dst + mat * DIM * DIM;
  const int idx = blockIdx.x * blockDim.x + threadIdx.x;   // 0..16383
  const int n = idx >> 7, k = idx & 127;
  d[n * DIM + k] = f2bf(s[k * DIM + n]);
}

// ---------------- zero fill (float4) ---------------------------------------
__global__ void zero_kernel(f32x4* __restrict__ p, int n4) {
  const int i = blockIdx.x * blockDim.x + threadIdx.x;
  if (i < n4) {
    const f32x4 z = {0.f, 0.f, 0.f, 0.f};
    p[i] = z;
  }
}

// ---------------- fused 2-layer MLP: out = relu(in@W1+b1)@W2+b2 -------------
__global__ __launch_bounds__(128) void mlp2_kernel(
    const float* __restrict__ in, float* __restrict__ out,
    const unsigned short* __restrict__ wt1, const float* __restrict__ b1,
    const unsigned short* __restrict__ wt2, const float* __restrict__ b2,
    int n) {
  __shared__ __align__(16) unsigned short ldsA[TROWS * LDAP];
  __shared__ __align__(16) unsigned short ldsB[TROWS * LDAP];
  const int tid  = threadIdx.x;
  const int row0 = blockIdx.x * TROWS;

  stage_tile(in, ldsA, row0, n);
  __syncthreads();

  const int lane  = tid & 31;
  const int row16 = (tid >> 5) * 16;
  const int nc0   = lane & 15;
  const int mof   = (lane < 16) ? 0 : 8;
  const v8f vzero = {0.f, 0.f, 0.f, 0.f, 0.f, 0.f, 0.f, 0.f};

  v8f acc[8];
#pragma unroll
  for (int i = 0; i < 8; ++i) acc[i] = vzero;
  gemm16x128(ldsA, row16, wt1, acc);

  // bias + ReLU -> LDS (bf16) for second layer
#pragma unroll
  for (int nt = 0; nt < 8; ++nt) {
    const int nc = nt * 16 + nc0;
    const float bb = b1[nc];
#pragma unroll
    for (int r = 0; r < 8; ++r) {
      float v = acc[nt][r] + bb;
      v = v > 0.f ? v : 0.f;
      ldsB[(row16 + mof + r) * LDAP + nc] = f2bf(v);
    }
  }
  __syncthreads();

  v8f acc2[8];
#pragma unroll
  for (int i = 0; i < 8; ++i) acc2[i] = vzero;
  gemm16x128(ldsB, row16, wt2, acc2);

#pragma unroll
  for (int nt = 0; nt < 8; ++nt) {
    const int nc = nt * 16 + nc0;
    const float bb = b2[nc];
#pragma unroll
    for (int r = 0; r < 8; ++r) {
      const int gr = row0 + row16 + mof + r;
      if (gr < n) out[gr * DIM + nc] = acc2[nt][r] + bb;
    }
  }
}

// ---------------- sparse COO spmm: agg[rows[e]] += vals[e]*x[cols[e]] -------
__global__ void spmm_edges(const int* __restrict__ rows, const int* __restrict__ cols,
                           const float* __restrict__ vals, const float* __restrict__ x,
                           float* __restrict__ agg, int E) {
  const int lane = threadIdx.x & 31;
  const int wid  = (blockIdx.x * blockDim.x + threadIdx.x) >> 5;
  const int nw   = (gridDim.x * blockDim.x) >> 5;
  for (int e = wid; e < E; e += nw) {
    const int   rr = rows[e];
    const int   cc = cols[e];
    const float vv = vals[e];
    const f32x4 xv = *(const f32x4*)(x + (size_t)cc * DIM + lane * 4);
    float* ap = agg + (size_t)rr * DIM + lane * 4;
    gfadd(ap + 0, vv * xv.x);
    gfadd(ap + 1, vv * xv.y);
    gfadd(ap + 2, vv * xv.z);
    gfadd(ap + 3, vv * xv.w);
  }
}

// ---------------- gates: z = sig(o@Wu1+x@Wu2+b), r = sig(...), c1 = o@Wo1+bo1
__global__ __launch_bounds__(128) void gates_kernel(
    const float* __restrict__ o, const float* __restrict__ x,
    const unsigned short* __restrict__ wtU1, const float* __restrict__ bu1,
    const unsigned short* __restrict__ wtU2, const float* __restrict__ bu2,
    const unsigned short* __restrict__ wtR1, const float* __restrict__ br1,
    const unsigned short* __restrict__ wtR2, const float* __restrict__ br2,
    const unsigned short* __restrict__ wtO1, const float* __restrict__ bo1,
    float* __restrict__ zb, float* __restrict__ rb, float* __restrict__ c1b,
    int n) {
  __shared__ __align__(16) unsigned short ldsO[TROWS * LDAP];
  __shared__ __align__(16) unsigned short ldsX[TROWS * LDAP];
  const int tid  = threadIdx.x;
  const int row0 = blockIdx.x * TROWS;

  stage_tile(o, ldsO, row0, n);
  stage_tile(x, ldsX, row0, n);
  __syncthreads();

  const int lane  = tid & 31;
  const int row16 = (tid >> 5) * 16;
  const int nc0   = lane & 15;
  const int mof   = (lane < 16) ? 0 : 8;
  const v8f vzero = {0.f, 0.f, 0.f, 0.f, 0.f, 0.f, 0.f, 0.f};
  v8f acc[8];

  // ---- z gate (two accumulated GEMMs) ----
#pragma unroll
  for (int i = 0; i < 8; ++i) acc[i] = vzero;
  gemm16x128(ldsO, row16, wtU1, acc);
  gemm16x128(ldsX, row16, wtU2, acc);
#pragma unroll
  for (int nt = 0; nt < 8; ++nt) {
    const int nc = nt * 16 + nc0;
    const float bb = bu1[nc] + bu2[nc];
#pragma unroll
    for (int r = 0; r < 8; ++r) {
      const int gr = row0 + row16 + mof + r;
      if (gr < n) zb[gr * DIM + nc] = sigmoidf_(acc[nt][r] + bb);
    }
  }

  // ---- r gate ----
#pragma unroll
  for (int i = 0; i < 8; ++i) acc[i] = vzero;
  gemm16x128(ldsO, row16, wtR1, acc);
  gemm16x128(ldsX, row16, wtR2, acc);
#pragma unroll
  for (int nt = 0; nt < 8; ++nt) {
    const int nc = nt * 16 + nc0;
    const float bb = br1[nc] + br2[nc];
#pragma unroll
    for (int r = 0; r < 8; ++r) {
      const int gr = row0 + row16 + mof + r;
      if (gr < n) rb[gr * DIM + nc] = sigmoidf_(acc[nt][r] + bb);
    }
  }

  // ---- c1 = o @ Wo1 + bo1 ----
#pragma unroll
  for (int i = 0; i < 8; ++i) acc[i] = vzero;
  gemm16x128(ldsO, row16, wtO1, acc);
#pragma unroll
  for (int nt = 0; nt < 8; ++nt) {
    const int nc = nt * 16 + nc0;
    const float bb = bo1[nc];
#pragma unroll
    for (int r = 0; r < 8; ++r) {
      const int gr = row0 + row16 + mof + r;
      if (gr < n) c1b[gr * DIM + nc] = acc[nt][r] + bb;
    }
  }
}

// ---------------- final: h = tanh(c1 + (r*x)@Wo2 + bo2); out=(1-z)x + z*h ---
__global__ __launch_bounds__(128) void final_kernel(
    const float* __restrict__ x, const float* __restrict__ rb,
    const float* __restrict__ zb, const float* __restrict__ c1b,
    const unsigned short* __restrict__ wtO2, const float* __restrict__ bo2,
    float* __restrict__ out, int n) {
  __shared__ __align__(16) unsigned short ldsRX[TROWS * LDAP];
  const int tid  = threadIdx.x;
  const int row0 = blockIdx.x * TROWS;

  // stage rx = r * x as bf16 (vectorized)
  for (int idx = tid; idx < TROWS * (DIM / 4); idx += 128) {
    const int r  = idx >> 5;
    const int c4 = (idx & 31) * 4;
    int gr = row0 + r; if (gr >= n) gr = n - 1;
    const size_t gi = (size_t)gr * DIM + c4;
    const f32x4 xv = *(const f32x4*)(x + gi);
    const f32x4 rv = *(const f32x4*)(rb + gi);
    f32x4 m;
    m.x = rv.x * xv.x; m.y = rv.y * xv.y; m.z = rv.z * xv.z; m.w = rv.w * xv.w;
    *(u32x2*)(ldsRX + r * LDAP + c4) = pack4bf(m);
  }
  __syncthreads();

  const int lane  = tid & 31;
  const int row16 = (tid >> 5) * 16;
  const int nc0   = lane & 15;
  const int mof   = (lane < 16) ? 0 : 8;
  const v8f vzero = {0.f, 0.f, 0.f, 0.f, 0.f, 0.f, 0.f, 0.f};
  v8f acc[8];
#pragma unroll
  for (int i = 0; i < 8; ++i) acc[i] = vzero;
  gemm16x128(ldsRX, row16, wtO2, acc);

#pragma unroll
  for (int nt = 0; nt < 8; ++nt) {
    const int nc = nt * 16 + nc0;
    const float bb = bo2[nc];
#pragma unroll
    for (int r = 0; r < 8; ++r) {
      const int gr = row0 + row16 + mof + r;
      if (gr < n) {
        const size_t gi = (size_t)gr * DIM + nc;
        const float h  = tanhf(acc[nt][r] + bb + c1b[gi]);
        const float zv = zb[gi];
        out[gi] = (1.0f - zv) * x[gi] + zv * h;
      }
    }
  }
}

// ---------------------------------------------------------------------------
extern "C" void kernel_launch(void* const* d_in, const int* in_sizes, int n_in,
                              void* d_out, int out_size, void* d_ws, size_t ws_size,
                              hipStream_t stream) {
  const float* x_in  = (const float*)d_in[0];
  const int*   rows  = (const int*)d_in[1];
  const int*   cols  = (const int*)d_in[2];
  const float* vals  = (const float*)d_in[3];
  const float* m1_W1 = (const float*)d_in[4];  const float* m1_b1 = (const float*)d_in[5];
  const float* m1_W2 = (const float*)d_in[6];  const float* m1_b2 = (const float*)d_in[7];
  const float* m2_W1 = (const float*)d_in[8];  const float* m2_b1 = (const float*)d_in[9];
  const float* m2_W2 = (const float*)d_in[10]; const float* m2_b2 = (const float*)d_in[11];
  const float* Wu1   = (const float*)d_in[12]; const float* bu1   = (const float*)d_in[13];
  const float* Wu2   = (const float*)d_in[14]; const float* bu2   = (const float*)d_in[15];
  const float* Wr1   = (const float*)d_in[16]; const float* br1   = (const float*)d_in[17];
  const float* Wr2   = (const float*)d_in[18]; const float* br2   = (const float*)d_in[19];
  const float* Wo1   = (const float*)d_in[20]; const float* bo1   = (const float*)d_in[21];
  const float* Wo2   = (const float*)d_in[22]; const float* bo2   = (const float*)d_in[23];

  const int N = in_sizes[0] / DIM;
  const int E = in_sizes[1];
  const size_t NB = (size_t)N * DIM * sizeof(float);
  const int DD = DIM * DIM;

  // workspace layout: [10x bf16 W^T][x][agg/out][z][r][c1]  (~128.3 MB)
  char* ws = (char*)d_ws;
  unsigned short* wt = (unsigned short*)ws;
  size_t off = (size_t)10 * DD * sizeof(unsigned short);  // 327680 (256-aligned)
  float* xb   = (float*)(ws + off); off += NB;
  float* aggb = (float*)(ws + off); off += NB;   // also reused as mlp2 output
  float* zb   = (float*)(ws + off); off += NB;
  float* rbuf = (float*)(ws + off); off += NB;
  float* c1b  = (float*)(ws + off); off += NB;

  // 1) weights -> bf16 transposed
  WPtrs wp = {{m1_W1, m1_W2, m2_W1, m2_W2, Wu1, Wu2, Wr1, Wr2, Wo1, Wo2}};
  wcvt_kernel<<<dim3(64, 10), 256, 0, stream>>>(wp, wt);

  const int gblocks = (N + TROWS - 1) / TROWS;

  // 2) x = mlp1(x_in)
  mlp2_kernel<<<gblocks, 128, 0, stream>>>(x_in, xb, wt + 0 * DD, m1_b1,
                                           wt + 1 * DD, m1_b2, N);

  // 3) agg = segment_sum(vals * x[cols], rows)
  zero_kernel<<<(N * DIM / 4 + 255) / 256, 256, 0, stream>>>((f32x4*)aggb, N * DIM / 4);
  spmm_edges<<<2048, 256, 0, stream>>>(rows, cols, vals, xb, aggb, E);

  // 4) out = mlp2(agg)   (in-place: each block reads its rows before writing)
  mlp2_kernel<<<gblocks, 128, 0, stream>>>(aggb, aggb, wt + 2 * DD, m2_b1,
                                           wt + 3 * DD, m2_b2, N);

  // 5) z, r, c1
  gates_kernel<<<gblocks, 128, 0, stream>>>(aggb, xb,
                                            wt + 4 * DD, bu1, wt + 5 * DD, bu2,
                                            wt + 6 * DD, br1, wt + 7 * DD, br2,
                                            wt + 8 * DD, bo1,
                                            zb, rbuf, c1b, N);

  // 6) h + blend -> d_out
  final_kernel<<<gblocks, 128, 0, stream>>>(xb, rbuf, zb, c1b,
                                            wt + 9 * DD, bo2, (float*)d_out, N);
}
